// EncoderBlock_3427383902900
// MI455X (gfx1250) — matrix operand order repaired and checked by reading, not compiled, not run on hardware
//
#include <hip/hip_runtime.h>

// ---------------------------------------------------------------------------
// MI455X (gfx1250) transformer encoder block.
// All matmuls on v_wmma_f32_16x16x32_bf16 (fp32 accumulate), wave32.
// K-tiles in flash attention staged into LDS by the Tensor Data Mover.
// ---------------------------------------------------------------------------

typedef __attribute__((ext_vector_type(16))) __bf16        v16bf;
typedef __attribute__((ext_vector_type(8)))  float         v8f;
typedef __attribute__((ext_vector_type(4)))  unsigned int  u32x4;
typedef __attribute__((ext_vector_type(4)))  int           i32x4;
typedef __attribute__((ext_vector_type(8)))  int           i32x8;

union Frag {
    v16bf v;
    uint4 q[2];
    unsigned short u[16];
};

__device__ __forceinline__ unsigned short f2bf(float f) {
    unsigned int u = __float_as_uint(f);
    u += 0x7FFFu + ((u >> 16) & 1u);        // round-to-nearest-even
    return (unsigned short)(u >> 16);
}

#define EB_DIM     1024
#define EB_HEADS   16
#define EB_HD      64
#define EB_SEQ     2048
#define EB_BATCH   4
#define EB_HIDDEN  4096
#define EB_MROWS   (EB_BATCH * EB_SEQ)      // 8192

// ---------------------------------------------------------------------------
// fp32 -> bf16 elementwise cast (weights)
// ---------------------------------------------------------------------------
__global__ __launch_bounds__(256) void cast_f32_bf16(const float* __restrict__ src,
                                                     unsigned short* __restrict__ dst,
                                                     int n) {
    int i = (blockIdx.x * 256 + threadIdx.x) * 4;
    if (i + 3 < n) {
        float4 v = *(const float4*)(src + i);
        dst[i + 0] = f2bf(v.x);
        dst[i + 1] = f2bf(v.y);
        dst[i + 2] = f2bf(v.z);
        dst[i + 3] = f2bf(v.w);
    }
}

// ---------------------------------------------------------------------------
// RMSNorm: one block per row of DIM=1024, 256 threads x 4 elements.
// ---------------------------------------------------------------------------
__global__ __launch_bounds__(256) void rmsnorm_bf16(const float* __restrict__ x,
                                                    const float* __restrict__ g,
                                                    unsigned short* __restrict__ out) {
    const int row  = blockIdx.x;
    const int lane = threadIdx.x & 31;
    const int wave = threadIdx.x >> 5;
    const float* xr = x + (size_t)row * EB_DIM;
    const int i0 = threadIdx.x * 4;

    float4 xv = *(const float4*)(xr + i0);
    float ss = xv.x * xv.x + xv.y * xv.y + xv.z * xv.z + xv.w * xv.w;
#pragma unroll
    for (int m = 16; m >= 1; m >>= 1)
        ss += __shfl_xor(ss, m, 32);

    __shared__ float sred[8];
    if (lane == 0) sred[wave] = ss;
    __syncthreads();
    float tot = 0.f;
#pragma unroll
    for (int w = 0; w < 8; ++w) tot += sred[w];

    const float rinv = rsqrtf(tot * (1.0f / EB_DIM) + 1e-6f);
    float4 gv = *(const float4*)(g + i0);
    unsigned short* op = out + (size_t)row * EB_DIM + i0;
    op[0] = f2bf(xv.x * rinv * gv.x);
    op[1] = f2bf(xv.y * rinv * gv.y);
    op[2] = f2bf(xv.z * rinv * gv.z);
    op[3] = f2bf(xv.w * rinv * gv.w);
}

// ---------------------------------------------------------------------------
// WMMA GEMM:  C[M,N] = A[M,K] @ W[N,K]^T + bias  (A,W bf16 row-major, K-major)
// Block: 128 thr = 4 waves (2x2), block tile 128x128.
// Wave tile 64x64 = 4x4 WMMA tiles: 16 wmma per 16 b128 loads per K-step
// (~32 FLOP/byte register-level intensity; weights stay L2-resident).
// ---------------------------------------------------------------------------
#define GM_QKV   0
#define GM_RESID 1
#define GM_RELU  2

template <int MODE, int KDIM, int NDIM>
__global__ __launch_bounds__(128) void gemm_bf16(const unsigned short* __restrict__ A,
                                                 const unsigned short* __restrict__ W,
                                                 const float* __restrict__ bias,
                                                 const float* __restrict__ resid,
                                                 float* __restrict__ outF,
                                                 unsigned short* __restrict__ outB) {
    const int lane = threadIdx.x & 31;
    const int wave = threadIdx.x >> 5;   // 0..3
    const int wm = wave & 1;
    const int wn = wave >> 1;
    const int hl  = lane >> 4;
    const int c16 = lane & 15;

    const int bm = blockIdx.y * 128 + wm * 64;
    const int bn = blockIdx.x * 128 + wn * 64;

    v8f acc[4][4];
    const v8f vzero = {0.f, 0.f, 0.f, 0.f, 0.f, 0.f, 0.f, 0.f};
#pragma unroll
    for (int i = 0; i < 4; ++i)
#pragma unroll
        for (int j = 0; j < 4; ++j) acc[i][j] = vzero;

    const int kbA = hl * 8;    // A frag: halves 0-7 -> kb.., 8-15 -> kb+16..
    const int kbB = hl * 8;    // B frag: same order as A: 0-7 -> kb.., 8-15 -> kb+16..

#pragma unroll 2
    for (int k0 = 0; k0 < KDIM; k0 += 32) {
        Frag a[4], b[4];
#pragma unroll
        for (int mt = 0; mt < 4; ++mt) {
            const unsigned short* p = A + (size_t)(bm + mt * 16 + c16) * KDIM + k0 + kbA;
            a[mt].q[0] = *(const uint4*)(p);
            a[mt].q[1] = *(const uint4*)(p + 16);
        }
#pragma unroll
        for (int nt = 0; nt < 4; ++nt) {
            const unsigned short* p = W + (size_t)(bn + nt * 16 + c16) * KDIM + k0 + kbB;
            b[nt].q[0] = *(const uint4*)(p);
            b[nt].q[1] = *(const uint4*)(p + 16);
        }
#pragma unroll
        for (int mt = 0; mt < 4; ++mt)
#pragma unroll
            for (int nt = 0; nt < 4; ++nt)
                acc[mt][nt] = __builtin_amdgcn_wmma_f32_16x16x32_bf16(
                    false, a[mt].v, false, b[nt].v, (short)0, acc[mt][nt], false, false);
    }

#pragma unroll
    for (int mt = 0; mt < 4; ++mt)
#pragma unroll
        for (int nt = 0; nt < 4; ++nt) {
            const int col = bn + nt * 16 + c16;
            const float bc = bias[col];
#pragma unroll
            for (int r = 0; r < 8; ++r) {
                const int row = bm + mt * 16 + r + 8 * hl;
                float v = acc[mt][nt][r] + bc;
                if (MODE == GM_QKV) {
                    const int bb = row >> 11;           // / SEQ
                    const int s  = row & (EB_SEQ - 1);
                    const int h  = col >> 6;            // / HD
                    const int d  = col & (EB_HD - 1);
                    outB[(((size_t)bb * EB_HEADS + h) * EB_SEQ + s) * EB_HD + d] = f2bf(v);
                } else if (MODE == GM_RESID) {
                    const size_t idx = (size_t)row * NDIM + col;
                    outF[idx] = resid[idx] + v;
                } else {  // GM_RELU
                    outB[(size_t)row * NDIM + col] = f2bf(v > 0.f ? v : 0.f);
                }
            }
        }
}

// ---------------------------------------------------------------------------
// Flash attention, bf16 WMMA, fp32 accumulate, TDM-staged K tiles.
// Grid (S/64, H, B), block 128 thr = 4 waves; each wave owns 16 q rows.
// K tile (64 keys x 64 dims) DMA'd into LDS by the Tensor Data Mover with
// descriptor padding (pad 4 dwords every 32 dwords) -> row stride 72 ushorts.
// V tile staged transposed (manual), P round-trips LDS for C->A layout.
// All B fragments are prefetched into register arrays before each WMMA burst
// so ds_load latency overlaps WMMA issue instead of draining DScnt per-op.
// ---------------------------------------------------------------------------
#define PSTR 72   // padded LDS row stride (multiple of 8 for 16B ds loads)

__global__ __launch_bounds__(128) void flash_attn(const unsigned short* __restrict__ Q,
                                                  const unsigned short* __restrict__ Km,
                                                  const unsigned short* __restrict__ V,
                                                  const int* __restrict__ emask,
                                                  unsigned short* __restrict__ O) {
    const int b = blockIdx.z, h = blockIdx.y, qt = blockIdx.x;
    const int lane = threadIdx.x & 31;
    const int wave = threadIdx.x >> 5;
    const int hl  = lane >> 4;
    const int c16 = lane & 15;

    const size_t headBase = ((size_t)(b * EB_HEADS + h)) * EB_SEQ * EB_HD;

    __shared__ __align__(16) unsigned short LK[64 * PSTR];   // K tile [key][d] padded
    __shared__ __align__(16) unsigned short VT[EB_HD * PSTR];// V tile transposed [d][key]
    __shared__ __align__(16) unsigned short Pb[4][16 * PSTR];// per-wave P tile [row][key]

    // --- Q fragments (A layout), loaded once: d-ksteps 0-31, 32-63 ---
    Frag aq[2];
    {
        const int m = qt * 64 + wave * 16 + c16;
        const unsigned short* qp = Q + headBase + (size_t)m * EB_HD + hl * 8;
        aq[0].q[0] = *(const uint4*)(qp);
        aq[0].q[1] = *(const uint4*)(qp + 16);
        aq[1].q[0] = *(const uint4*)(qp + 32);
        aq[1].q[1] = *(const uint4*)(qp + 48);
    }

    const v8f vzero = {0.f, 0.f, 0.f, 0.f, 0.f, 0.f, 0.f, 0.f};
    v8f o_acc[4];
    float row_m[8], row_l[8];
#pragma unroll
    for (int t = 0; t < 4; ++t) o_acc[t] = vzero;
#pragma unroll
    for (int r = 0; r < 8; ++r) { row_m[r] = -1e30f; row_l[r] = 0.f; }

    for (int j = 0; j < EB_SEQ; j += 64) {
        __syncthreads();   // previous iteration's readers of LK/VT are done

#if __has_builtin(__builtin_amdgcn_tensor_load_to_lds)
        // ---- TDM: DMA the 64x64 bf16 K tile into LDS (one issue per block).
        // D# pad: interval code 4 (32 dwords = one 128B row), amount code 3
        // (4 dwords = 16B) -> LDS row stride 144B = PSTR ushorts.
        if (wave == 0) {
            const unsigned lds_off =
                (unsigned)(unsigned long long)(const void*)&LK[0];
            const unsigned long long ga = (unsigned long long)(const void*)(
                Km + headBase + (size_t)j * EB_HD);
            u32x4 g0 = { 1u,                                  // count=1, user mode
                         lds_off,
                         (unsigned)(ga & 0xFFFFFFFFull),
                         (unsigned)((ga >> 32) & 0x01FFFFFFull) | (2u << 30) }; // type=2
            i32x8 g1 = { (int)((1u << 16)      // data_size = 2 bytes
                             | (1u << 20)      // pad_enable
                             | (4u << 22)      // pad_interval: 32 dwords
                             | (3u << 25)),    // pad_amount : 4 dwords
                         (int)(64u << 16),     // tensor_dim0 = 64 (lo16 in [31:16])
                         (int)(2048u << 16),   // tensor_dim1 = 2048 (lo16 in [31:16])
                         (int)(64u << 16),     // tile_dim0 = 64 ([31:16])
                         (int)64,              // tile_dim1 = 64, tile_dim2 = 0
                         (int)64,              // tensor_dim0_stride = 64 (lo32)
                         0, 0 };
            i32x4 gz = { 0, 0, 0, 0 };
#if defined(__clang_major__) && (__clang_major__ >= 23)
            i32x8 gz8 = { 0, 0, 0, 0, 0, 0, 0, 0 };
            __builtin_amdgcn_tensor_load_to_lds(g0, g1, gz, gz, gz8, 0);
#else
            __builtin_amdgcn_tensor_load_to_lds(g0, g1, gz, gz, 0);
#endif
#if __has_builtin(__builtin_amdgcn_s_wait_tensorcnt)
            __builtin_amdgcn_s_wait_tensorcnt(0);
#endif
        }
#else
        // ---- fallback: cooperative copy of K tile with padded stride
        for (int e = threadIdx.x; e < 64 * EB_HD; e += 128) {
            const int key = e >> 6;
            const int d   = e & 63;
            LK[key * PSTR + d] = Km[headBase + (size_t)(j + key) * EB_HD + d];
        }
#endif
        // ---- stage V tile transposed: VT[d][key] = V[j+key][d]
        for (int e = threadIdx.x; e < 64 * EB_HD; e += 128) {
            const int key = e >> 6;
            const int d   = e & 63;
            VT[d * PSTR + key] = V[headBase + (size_t)(j + key) * EB_HD + d];
        }
        __syncthreads();

        // --- S = Q @ K^T: prefetch all 8 K B-frags, then the WMMA burst
        Frag bk[4][2];
#pragma unroll
        for (int t = 0; t < 4; ++t)
#pragma unroll
            for (int ks = 0; ks < 2; ++ks) {
                const unsigned short* kp =
                    &LK[(t * 16 + c16) * PSTR + ks * 32 + hl * 8];
                bk[t][ks].q[0] = *(const uint4*)(kp);
                bk[t][ks].q[1] = *(const uint4*)(kp + 16);
            }
        v8f s[4];
#pragma unroll
        for (int t = 0; t < 4; ++t) {
            s[t] = vzero;
#pragma unroll
            for (int ks = 0; ks < 2; ++ks)
                s[t] = __builtin_amdgcn_wmma_f32_16x16x32_bf16(
                    false, aq[ks].v, false, bk[t][ks].v, (short)0, s[t], false, false);
        }

        // --- scale + mask (one key column per lane -> scalar mask per lane)
#pragma unroll
        for (int t = 0; t < 4; ++t) {
            const int key = j + t * 16 + c16;
            const bool ok = emask[b * EB_SEQ + key] != 0;
#pragma unroll
            for (int r = 0; r < 8; ++r)
                s[t][r] = ok ? s[t][r] * 0.125f : -1e9f;
        }

        // --- online softmax (rows live in 16-lane halves; xor masks <=8 stay in-half)
        float alpha[8];
#pragma unroll
        for (int r = 0; r < 8; ++r) {
            float mx = fmaxf(fmaxf(s[0][r], s[1][r]), fmaxf(s[2][r], s[3][r]));
#pragma unroll
            for (int m = 8; m >= 1; m >>= 1)
                mx = fmaxf(mx, __shfl_xor(mx, m, 32));
            const float nm = fmaxf(row_m[r], mx);
            alpha[r] = __expf(row_m[r] - nm);
            row_m[r] = nm;
        }
        float rs[8];
#pragma unroll
        for (int r = 0; r < 8; ++r) rs[r] = 0.f;
#pragma unroll
        for (int t = 0; t < 4; ++t)
#pragma unroll
            for (int r = 0; r < 8; ++r) {
                const float p = __expf(s[t][r] - row_m[r]);
                s[t][r] = p;
                rs[r] += p;
            }
#pragma unroll
        for (int r = 0; r < 8; ++r) {
#pragma unroll
            for (int m = 8; m >= 1; m >>= 1)
                rs[r] += __shfl_xor(rs[r], m, 32);
            row_l[r] = row_l[r] * alpha[r] + rs[r];
        }
#pragma unroll
        for (int t = 0; t < 4; ++t)
#pragma unroll
            for (int r = 0; r < 8; ++r)
                o_acc[t][r] = o_acc[t][r] * alpha[r];

        // --- P: C layout -> LDS -> A-fragment layout (same-wave, LDS in-order)
        unsigned short* pw = &Pb[wave][0];
#pragma unroll
        for (int t = 0; t < 4; ++t)
#pragma unroll
            for (int r = 0; r < 8; ++r)
                pw[(r + 8 * hl) * PSTR + t * 16 + c16] = f2bf(s[t][r]);

        // --- O += P @ V: prefetch P A-frags and all 8 V B-frags, then burst
        Frag pa[2];
#pragma unroll
        for (int ks = 0; ks < 2; ++ks) {
            const unsigned short* pp = pw + (size_t)c16 * PSTR + ks * 32 + hl * 8;
            pa[ks].q[0] = *(const uint4*)(pp);
            pa[ks].q[1] = *(const uint4*)(pp + 16);
        }
        Frag vb[4][2];
#pragma unroll
        for (int t = 0; t < 4; ++t)
#pragma unroll
            for (int ks = 0; ks < 2; ++ks) {
                const unsigned short* vp =
                    &VT[(t * 16 + c16) * PSTR + ks * 32 + hl * 8];
                vb[t][ks].q[0] = *(const uint4*)(vp);
                vb[t][ks].q[1] = *(const uint4*)(vp + 16);
            }
#pragma unroll
        for (int t = 0; t < 4; ++t)
#pragma unroll
            for (int ks = 0; ks < 2; ++ks)
                o_acc[t] = __builtin_amdgcn_wmma_f32_16x16x32_bf16(
                    false, pa[ks].v, false, vb[t][ks].v, (short)0, o_acc[t], false, false);
    }

    // --- epilogue: O / l, scatter to (B,S,DIM) bf16 ---
    float invl[8];
#pragma unroll
    for (int r = 0; r < 8; ++r) invl[r] = 1.0f / row_l[r];
#pragma unroll
    for (int t = 0; t < 4; ++t)
#pragma unroll
        for (int r = 0; r < 8; ++r) {
            const int m = qt * 64 + wave * 16 + r + 8 * hl;
            const size_t idx =
                ((size_t)b * EB_SEQ + m) * EB_DIM + h * EB_HD + t * 16 + c16;
            O[idx] = f2bf(o_acc[t][r] * invl[r]);
        }
}

// ---------------------------------------------------------------------------
// Workspace layout (bytes). Total required: ~136 MB.
// ---------------------------------------------------------------------------
static const size_t WQ_OFF = 0;
static const size_t WK_OFF = WQ_OFF + (size_t)EB_DIM * EB_DIM * 2;
static const size_t WV_OFF = WK_OFF + (size_t)EB_DIM * EB_DIM * 2;
static const size_t WO_OFF = WV_OFF + (size_t)EB_DIM * EB_DIM * 2;
static const size_t W1_OFF = WO_OFF + (size_t)EB_DIM * EB_DIM * 2;
static const size_t W2_OFF = W1_OFF + (size_t)EB_HIDDEN * EB_DIM * 2;
static const size_t XN_OFF = W2_OFF + (size_t)EB_HIDDEN * EB_DIM * 2;
static const size_t Q_OFF  = XN_OFF + (size_t)EB_MROWS * EB_DIM * 2;
static const size_t K_OFF  = Q_OFF  + (size_t)EB_MROWS * EB_DIM * 2;
static const size_t V_OFF  = K_OFF  + (size_t)EB_MROWS * EB_DIM * 2;
static const size_t O_OFF  = V_OFF  + (size_t)EB_MROWS * EB_DIM * 2;
static const size_t X2_OFF = O_OFF  + (size_t)EB_MROWS * EB_DIM * 2;
static const size_t H_OFF  = Q_OFF;   // reuse dead q/k/v/o region (exactly 64 MB)

extern "C" void kernel_launch(void* const* d_in, const int* in_sizes, int n_in,
                              void* d_out, int out_size, void* d_ws, size_t ws_size,
                              hipStream_t stream) {
    const float* x      = (const float*)d_in[0];
    const int*   emask  = (const int*)  d_in[1];
    const float* wq     = (const float*)d_in[2];
    const float* bq     = (const float*)d_in[3];
    const float* wk     = (const float*)d_in[4];
    const float* bk     = (const float*)d_in[5];
    const float* wv     = (const float*)d_in[6];
    const float* bv     = (const float*)d_in[7];
    const float* wo     = (const float*)d_in[8];
    const float* bo     = (const float*)d_in[9];
    const float* w1     = (const float*)d_in[10];
    const float* b1     = (const float*)d_in[11];
    const float* w2     = (const float*)d_in[12];
    const float* b2     = (const float*)d_in[13];
    const float* g_attn = (const float*)d_in[14];
    const float* g_ffn  = (const float*)d_in[15];

    char* ws = (char*)d_ws;
    unsigned short* wq_b = (unsigned short*)(ws + WQ_OFF);
    unsigned short* wk_b = (unsigned short*)(ws + WK_OFF);
    unsigned short* wv_b = (unsigned short*)(ws + WV_OFF);
    unsigned short* wo_b = (unsigned short*)(ws + WO_OFF);
    unsigned short* w1_b = (unsigned short*)(ws + W1_OFF);
    unsigned short* w2_b = (unsigned short*)(ws + W2_OFF);
    unsigned short* xn_b = (unsigned short*)(ws + XN_OFF);
    unsigned short* q_b  = (unsigned short*)(ws + Q_OFF);
    unsigned short* k_b  = (unsigned short*)(ws + K_OFF);
    unsigned short* v_b  = (unsigned short*)(ws + V_OFF);
    unsigned short* o_b  = (unsigned short*)(ws + O_OFF);
    float*          x2   = (float*)        (ws + X2_OFF);
    unsigned short* h_b  = (unsigned short*)(ws + H_OFF);

    const int nSmall = EB_DIM * EB_DIM;       // 1M
    const int nBig   = EB_HIDDEN * EB_DIM;    // 4M
    cast_f32_bf16<<<nSmall / 1024, 256, 0, stream>>>(wq, wq_b, nSmall);
    cast_f32_bf16<<<nSmall / 1024, 256, 0, stream>>>(wk, wk_b, nSmall);
    cast_f32_bf16<<<nSmall / 1024, 256, 0, stream>>>(wv, wv_b, nSmall);
    cast_f32_bf16<<<nSmall / 1024, 256, 0, stream>>>(wo, wo_b, nSmall);
    cast_f32_bf16<<<nBig   / 1024, 256, 0, stream>>>(w1, w1_b, nBig);
    cast_f32_bf16<<<nBig   / 1024, 256, 0, stream>>>(w2, w2_b, nBig);

    // xn = rmsnorm(x, g_attn)
    rmsnorm_bf16<<<EB_MROWS, 256, 0, stream>>>(x, g_attn, xn_b);

    // Q/K/V projections, scattered to (B,H,S,HD)
    const dim3 gProj(EB_DIM / 128, EB_MROWS / 128);
    gemm_bf16<GM_QKV, EB_DIM, EB_DIM><<<gProj, 128, 0, stream>>>(
        xn_b, wq_b, bq, nullptr, nullptr, q_b);
    gemm_bf16<GM_QKV, EB_DIM, EB_DIM><<<gProj, 128, 0, stream>>>(
        xn_b, wk_b, bk, nullptr, nullptr, k_b);
    gemm_bf16<GM_QKV, EB_DIM, EB_DIM><<<gProj, 128, 0, stream>>>(
        xn_b, wv_b, bv, nullptr, nullptr, v_b);

    // flash attention -> o (B,S,DIM)
    flash_attn<<<dim3(EB_SEQ / 64, EB_HEADS, EB_BATCH), 128, 0, stream>>>(
        q_b, k_b, v_b, emask, o_b);

    // x2 = x + o @ wo^T + bo
    gemm_bf16<GM_RESID, EB_DIM, EB_DIM><<<gProj, 128, 0, stream>>>(
        o_b, wo_b, bo, x, x2, nullptr);

    // xn = rmsnorm(x2, g_ffn)
    rmsnorm_bf16<<<EB_MROWS, 256, 0, stream>>>(x2, g_ffn, xn_b);

    // h = relu(xn @ w1^T + b1)
    gemm_bf16<GM_RELU, EB_DIM, EB_HIDDEN><<<dim3(EB_HIDDEN / 128, EB_MROWS / 128),
                                            128, 0, stream>>>(
        xn_b, w1_b, b1, nullptr, nullptr, h_b);

    // out = x2 + h @ w2^T + b2
    gemm_bf16<GM_RESID, EB_HIDDEN, EB_DIM><<<gProj, 128, 0, stream>>>(
        h_b, w2_b, b2, x2, (float*)d_out, nullptr);
}
